// FullPotts_47442208751686
// MI455X (gfx1250) — matrix operand order repaired and checked by reading, not compiled, not run on hardware
//
#include <hip/hip_runtime.h>
#include <hip/hip_bf16.h>

typedef __attribute__((ext_vector_type(16))) __bf16 v16bf;
typedef __attribute__((ext_vector_type(8)))  float  v8f;

#define LL 256          // L
#define MM 512          // M
#define AA 21           // A
#define KK (LL*AA)      // 5376 = K of gemm1, also N of gemm1

#define BM 128
#define BN 128
#define BK 32
#define ASTR 40         // A-tile LDS stride (bf16); 80B rows -> 16B aligned
#define BPTSTR 20       // B-tile packed-pair-transposed stride (uints); 80B rows

union FragBF { v16bf v; unsigned u[8]; };
union PackBF { unsigned u; __bf16 h[2]; };

__device__ __forceinline__ void async_b128(unsigned lds, unsigned long long ga)
{
  asm volatile("global_load_async_to_lds_b128 %0, %1, off"
               :: "v"(lds), "v"(ga) : "memory");
}
__device__ __forceinline__ void wait_async0()
{
  asm volatile("s_wait_asynccnt 0x0" ::: "memory");
}

// ---------------------------------------------------------------------------
// prep: Xbf[m, l*21+a] = bf16(MSA[l,m,a]);  Xwbf = bf16(MSA[l,m,a]*lw[l])
// ---------------------------------------------------------------------------
__global__ __launch_bounds__(256) void prep_kernel(
    const float* __restrict__ MSA, const float* __restrict__ lw,
    __bf16* __restrict__ Xbf, __bf16* __restrict__ Xwbf)
{
  int idx = blockIdx.x * 256 + threadIdx.x;      // < MM*KK (exact)
  int m   = idx / KK;
  int rem = idx - m * KK;                        // l*21 + a
  int l   = rem / AA;
  int a   = rem - l * AA;
  float v = MSA[(l * MM + m) * AA + a];
  Xbf[idx]  = (__bf16)v;
  Xwbf[idx] = (__bf16)(v * lw[l]);
}

// ---------------------------------------------------------------------------
// GEMM1: Hedge[m, l*21+c] = sum_{p,a} Xbf[m, p*21+a] * J[l,p,c,a]
// A tile via async global->LDS b128; B tile gathered from f32 J, bf16-pair
// packed into transposed uint LDS rows (Bpt[n][kp]).
// ---------------------------------------------------------------------------
__global__ __launch_bounds__(256) void gemm_edge_kernel(
    const __bf16* __restrict__ X, const float* __restrict__ J,
    float* __restrict__ C)
{
  __shared__ __bf16 As[BM * ASTR];
  __shared__ unsigned Bpt[BN * BPTSTR];   // Bpt[n][kp] = {K=2kp, 2kp+1} at col n

  const int tid  = threadIdx.x;
  const int lane = tid & 31;
  const int wv   = tid >> 5;               // wave id 0..7
  const int wm   = (wv & 1) * 64;          // wave M offset in block tile
  const int wn   = (wv >> 1) * 32;         // wave N offset
  const int row  = lane & 15;
  const int khA  = (lane >> 4) << 4;       // A: hi lanes hold K+16
  const int kpB  = (lane >> 4) << 3;       // B: hi lanes hold pairs 8..15 (K+16)

  const int n0 = blockIdx.x * BN;
  const int m0 = blockIdx.y * BM;

  const unsigned asBase = (unsigned)(unsigned long long)(void*)As;

  v8f zero = {};
  v8f acc[4][2];
  #pragma unroll
  for (int i = 0; i < 4; ++i)
    #pragma unroll
    for (int j = 0; j < 2; ++j) acc[i][j] = zero;

  // A loader coords: thread -> (row, 16-col half)
  const int aR = tid >> 1;                 // 0..127
  const int aK = (tid & 1) << 4;           // 0 or 16
  const unsigned aLds = asBase + (unsigned)(aR * ASTR + aK) * 2u;

  // B loader coords: thread -> (pair-row kp, 8 cols)
  const int bKp = tid >> 4;                // 0..15
  const int bN  = (tid & 15) << 3;         // 0..120

  for (int kt = 0; kt < KK; kt += BK) {
    // ---- A tile: two 16B async copies per thread ----
    {
      unsigned long long ga =
          (unsigned long long)(const void*)(X + (m0 + aR) * KK + kt + aK);
      async_b128(aLds, ga);
      async_b128(aLds + 16u, ga + 16ull);
    }
    // ---- B tile: gather J (f32), convert, pack pairs into Bpt[n][kp] ----
    {
      int k0 = kt + 2 * bKp;
      int p0 = k0 / AA, a0 = k0 - p0 * AA;
      int k1 = k0 + 1;
      int p1 = k1 / AA, a1 = k1 - p1 * AA;
      long base0 = (long)p0 * (AA * AA) + a0;
      long base1 = (long)p1 * (AA * AA) + a1;
      #pragma unroll
      for (int i = 0; i < 8; ++i) {
        int nn = n0 + bN + i;              // global n = l*21 + c
        int l  = nn / AA;
        int c  = nn - l * AA;
        long jb = (long)l * (LL * AA * AA) + (long)c * AA;
        PackBF pk;
        pk.h[0] = (__bf16)J[jb + base0];
        pk.h[1] = (__bf16)J[jb + base1];
        Bpt[(bN + i) * BPTSTR + bKp] = pk.u;
      }
    }
    wait_async0();
    __syncthreads();

    // ---- fragments + WMMA ----
    FragBF bfr[2];
    #pragma unroll
    for (int ni = 0; ni < 2; ++ni) {
      const unsigned* Br = Bpt + (wn + ni * 16 + row) * BPTSTR + kpB;
      #pragma unroll
      for (int v = 0; v < 8; ++v) bfr[ni].u[v] = Br[v];
    }
    #pragma unroll
    for (int mi = 0; mi < 4; ++mi) {
      FragBF af;
      const __bf16* Ar = As + (wm + mi * 16 + row) * ASTR;
      #pragma unroll
      for (int v = 0; v < 8; ++v) {
        int k = (v << 1) + khA;
        af.u[v] = *(const unsigned*)(Ar + k);
      }
      acc[mi][0] = __builtin_amdgcn_wmma_f32_16x16x32_bf16(
          false, af.v, false, bfr[0].v, (short)0, acc[mi][0], false, false);
      acc[mi][1] = __builtin_amdgcn_wmma_f32_16x16x32_bf16(
          false, af.v, false, bfr[1].v, (short)0, acc[mi][1], false, false);
    }
    __syncthreads();
  }

  // ---- store D (f32 16x16 layout: VGPR r -> row r (+8 for hi lanes)) ----
  #pragma unroll
  for (int mi = 0; mi < 4; ++mi)
    #pragma unroll
    for (int ni = 0; ni < 2; ++ni) {
      int ng = n0 + wn + ni * 16 + (lane & 15);
      int mb = m0 + wm + mi * 16 + ((lane >> 4) << 3);
      #pragma unroll
      for (int r = 0; r < 8; ++r)
        C[(long)(mb + r) * KK + ng] = acc[mi][ni][r];
    }
}

// ---------------------------------------------------------------------------
// GEMM2: simU[m,n] = sum_k Xwbf[m,k] * Xbf[n,k]   (512x512, K=5376)
// Both tiles fed entirely by async global->LDS b128 copies.
// ---------------------------------------------------------------------------
__global__ __launch_bounds__(256) void gemm_gram_kernel(
    const __bf16* __restrict__ Xw, const __bf16* __restrict__ X,
    float* __restrict__ simU)
{
  __shared__ __bf16 As[BM * ASTR];
  __shared__ unsigned Bpt[BN * BPTSTR];

  const int tid  = threadIdx.x;
  const int lane = tid & 31;
  const int wv   = tid >> 5;
  const int wm   = (wv & 1) * 64;
  const int wn   = (wv >> 1) * 32;
  const int row  = lane & 15;
  const int khA  = (lane >> 4) << 4;
  const int kpB  = (lane >> 4) << 3;

  const int n0 = blockIdx.x * BN;
  const int m0 = blockIdx.y * BM;

  const unsigned asBase  = (unsigned)(unsigned long long)(void*)As;
  const unsigned bptBase = (unsigned)(unsigned long long)(void*)Bpt;

  v8f zero = {};
  v8f acc[4][2];
  #pragma unroll
  for (int i = 0; i < 4; ++i)
    #pragma unroll
    for (int j = 0; j < 2; ++j) acc[i][j] = zero;

  const int aR = tid >> 1;                 // 0..127
  const int aK = (tid & 1) << 4;           // 0 or 16
  const unsigned aLds = asBase + (unsigned)(aR * ASTR + aK) * 2u;
  const int bKp0 = (tid & 1) << 3;         // pair base 0 or 8
  const unsigned bLds = bptBase + (unsigned)(aR * BPTSTR + bKp0) * 4u;

  for (int kt = 0; kt < KK; kt += BK) {
    // ---- A tile ----
    {
      unsigned long long ga =
          (unsigned long long)(const void*)(Xw + (m0 + aR) * KK + kt + aK);
      async_b128(aLds, ga);
      async_b128(aLds + 16u, ga + 16ull);
    }
    // ---- B tile: row n = aR of X, 16 contiguous k -> 8 consecutive pairs ----
    {
      unsigned long long ga =
          (unsigned long long)(const void*)(X + (n0 + aR) * KK + kt + aK);
      async_b128(bLds, ga);
      async_b128(bLds + 16u, ga + 16ull);
    }
    wait_async0();
    __syncthreads();

    FragBF bfr[2];
    #pragma unroll
    for (int ni = 0; ni < 2; ++ni) {
      const unsigned* Br = Bpt + (wn + ni * 16 + row) * BPTSTR + kpB;
      #pragma unroll
      for (int v = 0; v < 8; ++v) bfr[ni].u[v] = Br[v];
    }
    #pragma unroll
    for (int mi = 0; mi < 4; ++mi) {
      FragBF af;
      const __bf16* Ar = As + (wm + mi * 16 + row) * ASTR;
      #pragma unroll
      for (int v = 0; v < 8; ++v) {
        int k = (v << 1) + khA;
        af.u[v] = *(const unsigned*)(Ar + k);
      }
      acc[mi][0] = __builtin_amdgcn_wmma_f32_16x16x32_bf16(
          false, af.v, false, bfr[0].v, (short)0, acc[mi][0], false, false);
      acc[mi][1] = __builtin_amdgcn_wmma_f32_16x16x32_bf16(
          false, af.v, false, bfr[1].v, (short)0, acc[mi][1], false, false);
    }
    __syncthreads();
  }

  #pragma unroll
  for (int mi = 0; mi < 4; ++mi)
    #pragma unroll
    for (int ni = 0; ni < 2; ++ni) {
      int ng = n0 + wn + ni * 16 + (lane & 15);
      int mb = m0 + wm + mi * 16 + ((lane >> 4) << 3);
      #pragma unroll
      for (int r = 0; r < 8; ++r)
        simU[(mb + r) * MM + ng] = acc[mi][ni][r];
    }
}

// ---------------------------------------------------------------------------
// per (m,l): node term, edge_wt, H_diff, log_softmax(20), logp_msa; Hout for m=0
// ---------------------------------------------------------------------------
__global__ __launch_bounds__(256) void node_logp_kernel(
    const float* __restrict__ MSA, const float* __restrict__ h,
    const float* __restrict__ Hedge, float* __restrict__ logp_msa,
    float* __restrict__ Hout)
{
  int m = blockIdx.x;      // 0..511
  int l = threadIdx.x;     // 0..255
  const float* msa = MSA + (l * MM + m) * AA;
  const float* hv  = h + l * AA;
  const float* He  = Hedge + (long)m * KK + l * AA;

  float mv[AA], he[AA], Hd[AA];
  float node_wt = 0.f, edge_wt = 0.f;
  #pragma unroll
  for (int a = 0; a < AA; ++a) { mv[a] = msa[a]; node_wt += mv[a] * hv[a]; }
  #pragma unroll
  for (int c = 0; c < AA; ++c) { he[c] = He[c]; edge_wt += he[c] * mv[c]; }
  #pragma unroll
  for (int c = 0; c < AA; ++c) Hd[c] = hv[c] - node_wt + he[c] - edge_wt;

  float mx = -1e30f;
  #pragma unroll
  for (int c = 0; c < 20; ++c) mx = fmaxf(mx, -Hd[c]);
  float se = 0.f;
  #pragma unroll
  for (int c = 0; c < 20; ++c) se += expf(-Hd[c] - mx);
  float lse = mx + logf(se);
  float lp = 0.f;
  #pragma unroll
  for (int c = 0; c < 20; ++c) lp += (-Hd[c] - lse) * mv[c];
  logp_msa[m * LL + l] = lp;

  if (m == 0) {
    #pragma unroll
    for (int c = 0; c < 20; ++c) Hout[l * 20 + c] = Hd[c];
  }
}

__device__ inline float blockReduce256(float* red, int t, float v)
{
  red[t] = v;
  #pragma unroll
  for (int s = 128; s > 0; s >>= 1) {
    __syncthreads();
    if (t < s) red[t] += red[t + s];
  }
  __syncthreads();
  float r = red[0];
  __syncthreads();
  return r;
}

// ---------------------------------------------------------------------------
// per m: lp[m] = sum_l logp_msa*lw ; simw[m] = dw[m] / sum_n sigmoid(...)
// ---------------------------------------------------------------------------
__global__ __launch_bounds__(256) void perm_kernel(
    const float* __restrict__ logp_msa, const float* __restrict__ lw,
    const float* __restrict__ simU, const float* __restrict__ dw,
    float* __restrict__ lp, float* __restrict__ simw)
{
  __shared__ float red[256];
  int m = blockIdx.x, t = threadIdx.x;

  float sumLW = blockReduce256(red, t, lw[t]);
  float lpm   = blockReduce256(red, t, logp_msa[m * LL + t] * lw[t]);

  float inv = 1.f / sumLW;
  float s0 = simU[m * MM + t] * inv;
  float s1 = simU[m * MM + 256 + t] * inv;
  float term = 1.f / (1.f + expf(-10.f * (s0 - 0.8f))) +
               1.f / (1.f + expf(-10.f * (s1 - 0.8f)));
  float ssum = blockReduce256(red, t, term);

  if (t == 0) { lp[m] = lpm; simw[m] = dw[m] / ssum; }
}

// ---------------------------------------------------------------------------
// final: logp = sum(simw*lp)/sum(simw) -> d_out[5120]
// ---------------------------------------------------------------------------
__global__ __launch_bounds__(512) void final_kernel(
    const float* __restrict__ lp, const float* __restrict__ simw,
    float* __restrict__ out)
{
  __shared__ float red[512];
  int t = threadIdx.x;
  float w = simw[t];

  red[t] = w * lp[t];
  #pragma unroll
  for (int s = 256; s > 0; s >>= 1) { __syncthreads(); if (t < s) red[t] += red[t + s]; }
  __syncthreads();
  float num = red[0];
  __syncthreads();

  red[t] = w;
  #pragma unroll
  for (int s = 256; s > 0; s >>= 1) { __syncthreads(); if (t < s) red[t] += red[t + s]; }
  __syncthreads();
  if (t == 0) out[LL * 20] = num / red[0];
}

extern "C" void kernel_launch(void* const* d_in, const int* in_sizes, int n_in,
                              void* d_out, int out_size, void* d_ws, size_t ws_size,
                              hipStream_t stream)
{
  (void)in_sizes; (void)n_in; (void)out_size; (void)ws_size;
  const float* MSA = (const float*)d_in[0];
  const float* h   = (const float*)d_in[1];
  const float* J   = (const float*)d_in[2];
  const float* lw  = (const float*)d_in[3];
  const float* dw  = (const float*)d_in[4];
  float* out = (float*)d_out;

  char* ws = (char*)d_ws;
  __bf16* Xbf   = (__bf16*)(ws);                 //  5,505,024 B
  __bf16* Xwbf  = (__bf16*)(ws + 5505024);       //  5,505,024 B
  float*  Hedge = (float*)(ws + 11010048);       // 11,010,048 B
  float*  simU  = (float*)(ws + 22020096);       //  1,048,576 B
  float*  lpm   = (float*)(ws + 23068672);       //    524,288 B
  float*  lpv   = (float*)(ws + 23592960);       //      2,048 B
  float*  swv   = (float*)(ws + 23595008);       //      2,048 B

  prep_kernel<<<(MM * KK) / 256, 256, 0, stream>>>(MSA, lw, Xbf, Xwbf);
  gemm_edge_kernel<<<dim3(KK / BN, MM / BM), 256, 0, stream>>>(Xbf, J, Hedge);
  gemm_gram_kernel<<<dim3(MM / BN, MM / BM), 256, 0, stream>>>(Xwbf, Xbf, simU);
  node_logp_kernel<<<MM, LL, 0, stream>>>(MSA, h, Hedge, lpm, out);
  perm_kernel<<<MM, 256, 0, stream>>>(lpm, lw, simU, dw, lpv, swv);
  final_kernel<<<1, 512, 0, stream>>>(lpv, swv, out);
}
